// Mamba2Block_76390288326774
// MI455X (gfx1250) — matrix operand order repaired and checked by reading, not compiled, not run on hardware
//
#include <hip/hip_runtime.h>
#include <stdint.h>

// ---------------------------------------------------------------------------
// Mamba2 block for MI455X (gfx1250, wave32, WMMA + async global->LDS).
//   zxbca = x @ W_in^T           -> bf16 WMMA GEMM (compute-bound, 276 GFLOP)
//   selective-scan over L        -> 64 blocks (B*H), state in registers
//   y = scan * silu(z); LN       -> fused elementwise + reduction, emit bf16
//   out = y_ln @ W_out^T         -> bf16 WMMA GEMM
// GEMM tiles are staged with GLOBAL_LOAD_ASYNC_TO_LDS_B128 (ASYNCcnt path),
// double-buffered in LDS so the DMA overlaps the v_wmma issue.
// ---------------------------------------------------------------------------

#define BB      2
#define LL      2048
#define DM      2048
#define HH      32
#define PP      64
#define NN      64
#define DI      2048              // H*P
#define PROJ    8224              // 2*DI + 2*H*N + H
#define NPAD1   8320              // 65 * 128  (PROJ padded to N-tile multiple)
#define MROWS   (BB * LL)         // 8192
#define EPS     1e-5f

typedef __attribute__((ext_vector_type(16))) __bf16 v16bf;
typedef __attribute__((ext_vector_type(8)))  float  v8f;

union Frag {                      // 16 bf16 = 8 VGPRs = 2x uint4
    v16bf v;
    uint4 q[2];
};

// async global->LDS staging (inline asm: builtins differ across toolchains).
// INST_OFFSET is added to BOTH the global and the LDS address (ISA 10.7.4),
// so one base pair + offset:16 covers each thread's 32-byte strip.
#define ASYNC_LD16(ldsoff, gaddr)                                             \
    asm volatile("global_load_async_to_lds_b128 %0, %1, off"                  \
                 :: "v"(ldsoff), "v"(gaddr) : "memory")
#define ASYNC_LD16_HI(ldsoff, gaddr)                                          \
    asm volatile("global_load_async_to_lds_b128 %0, %1, off offset:16"        \
                 :: "v"(ldsoff), "v"(gaddr) : "memory")
#define WAIT_ASYNC0()                                                         \
    asm volatile("s_wait_asynccnt 0x0" ::: "memory")

// ----------------------------- helpers -------------------------------------

__device__ __forceinline__ uint16_t f2bf(float f) {
    uint32_t u = __float_as_uint(f);
    if ((u & 0x7fffffffu) > 0x7f800000u) return (uint16_t)((u >> 16) | 0x40u); // qNaN
    u += 0x7fffu + ((u >> 16) & 1u);                                           // RNE
    return (uint16_t)(u >> 16);
}

// f32 -> bf16 with zero-padding past nreal (for the padded W_in copy)
__global__ void cvt_f32_bf16(const float* __restrict__ in, uint16_t* __restrict__ out,
                             long nreal, long ntot) {
    long stride = (long)gridDim.x * blockDim.x;
    for (long i = (long)blockIdx.x * blockDim.x + threadIdx.x; i < ntot; i += stride)
        out[i] = (i < nreal) ? f2bf(in[i]) : (uint16_t)0;
}

// ----------------------------- WMMA GEMM -----------------------------------
// C[M x Nreal] (f32, ld = ldC) = A[M x K] (bf16) * Bm[Npad x K]^T (bf16)
// Block tile 128x128, K-tile 32, 8 waves: 4 (M) x 2 (N), wave tile 32x64.
// Double-buffered LDS, async DMA staging.

#define TILE_M 128
#define TILE_N 128
#define TILE_K 32
#define LDSS   40                  // bf16 per LDS row: 32 data + 8 pad (80 B)
#define BUF_ELT (2 * TILE_M * LDSS)   // 10240 uint16 per buffer (A + B)
#define BUF_BYTES (BUF_ELT * 2)       // 20480 B

__global__ __launch_bounds__(256)
void wmma_gemm_bf16(const uint16_t* __restrict__ A,
                    const uint16_t* __restrict__ Bm,
                    float* __restrict__ C,
                    int K, int Nreal, int ldC) {
    __shared__ __align__(16) uint16_t lds[2 * BUF_ELT];   // 40 KB, 2 buffers

    const int t    = threadIdx.x;
    const int lane = t & 31;
    const int wid  = t >> 5;
    const int wm   = wid & 3;              // 0..3  -> 32-row strip
    const int wn   = wid >> 2;             // 0..1  -> 64-col strip
    const int blockM = blockIdx.y * TILE_M;
    const int blockN = blockIdx.x * TILE_N;

    // cooperative tile load mapping: row = t>>1 (0..127), k-offset = (t&1)*16
    const int ldRow = t >> 1;
    const int ldK   = (t & 1) * 16;

    // wave-relative LDS byte offsets for this thread's async destinations
    const uint32_t ldsBase = (uint32_t)(uintptr_t)(void*)lds;
    const uint32_t dstA0   = ldsBase + (uint32_t)(ldRow * LDSS + ldK) * 2u;
    // global byte addresses for this thread's strips
    const uint64_t gAbase = (uint64_t)(uintptr_t)(A  + (size_t)(blockM + ldRow) * K + ldK);
    const uint64_t gBbase = (uint64_t)(uintptr_t)(Bm + (size_t)(blockN + ldRow) * K + ldK);

    v8f acc[2][4];
#pragma unroll
    for (int mi = 0; mi < 2; ++mi)
#pragma unroll
        for (int ni = 0; ni < 4; ++ni)
            acc[mi][ni] = (v8f){0.f, 0.f, 0.f, 0.f, 0.f, 0.f, 0.f, 0.f};

    // per-lane fragment decode (ISA 7.12.2, 16-bit A 16x32 / B 32x16)
    const int fr   = lane & 15;            // A: M row   B: N col
    const int kh8  = (lane >> 4) << 3;     // A K-half select {0,8}
    const int kh16 = (lane >> 4) << 4;     // unused: B also uses kh8 {0,8}

    const int nk = K / TILE_K;

    // issue one k-tile's async DMA into buffer (kt & 1)
    auto issue_stage = [&](int kt) {
        const uint32_t bo = (uint32_t)(kt & 1) * BUF_BYTES;
        const uint32_t la = dstA0 + bo;
        const uint32_t lb = la + (uint32_t)(TILE_M * LDSS * 2);   // +10240 B
        const uint64_t ga = gAbase + (uint64_t)kt * (TILE_K * 2);
        const uint64_t gb = gBbase + (uint64_t)kt * (TILE_K * 2);
        ASYNC_LD16(la, ga);  ASYNC_LD16_HI(la, ga);
        ASYNC_LD16(lb, gb);  ASYNC_LD16_HI(lb, gb);
    };

    issue_stage(0);

    for (int kt = 0; kt < nk; ++kt) {
        WAIT_ASYNC0();        // this wave's DMA for buffer kt landed
        __syncthreads();      // all waves' DMA landed; all waves done with buf kt^1
        if (kt + 1 < nk) issue_stage(kt + 1);

        const uint16_t* bufA = lds + (kt & 1) * BUF_ELT;
        const uint16_t* bufB = bufA + TILE_M * LDSS;

        Frag af[2], bf[4];
#pragma unroll
        for (int mi = 0; mi < 2; ++mi) {
            const uint16_t* p = bufA + (wm * 32 + mi * 16 + fr) * LDSS + kh8;
            af[mi].q[0] = *(const uint4*)(p);        // K = kh8 .. kh8+7
            af[mi].q[1] = *(const uint4*)(p + 16);   // K = 16+kh8 .. +7
        }
#pragma unroll
        for (int ni = 0; ni < 4; ++ni) {
            const uint16_t* p = bufB + (wn * 64 + ni * 16 + fr) * LDSS + kh8;
            bf[ni].q[0] = *(const uint4*)(p);        // K = kh8 .. kh8+7
            bf[ni].q[1] = *(const uint4*)(p + 16);   // K = 16+kh8 .. +7
        }
#pragma unroll
        for (int mi = 0; mi < 2; ++mi)
#pragma unroll
            for (int ni = 0; ni < 4; ++ni)
                acc[mi][ni] = __builtin_amdgcn_wmma_f32_16x16x32_bf16(
                    false, af[mi].v, false, bf[ni].v,
                    (short)0, acc[mi][ni], false, false);
    }

    // epilogue: C/D layout -> VGPR r, lanes 0-15: M=r, lanes 16-31: M=r+8
    const int mofs = (lane >> 4) << 3;
#pragma unroll
    for (int mi = 0; mi < 2; ++mi)
#pragma unroll
        for (int ni = 0; ni < 4; ++ni) {
            int col = blockN + wn * 64 + ni * 16 + (lane & 15);
            if (col < Nreal) {
#pragma unroll
                for (int r = 0; r < 8; ++r) {
                    int row = blockM + wm * 32 + mi * 16 + r + mofs;
                    C[(size_t)row * ldC + col] = acc[mi][ni][r];
                }
            }
        }
}

// ----------------------------- SSM scan ------------------------------------
// One block per (b, h). 256 threads: thread t owns p = t>>2, n-range (t&3)*16.
// State S[p][16] lives in registers; b/c/x/decay staged through LDS per step.

__global__ __launch_bounds__(256)
void scan_kernel(const float* __restrict__ zxbca, const float* __restrict__ Dmat,
                 float* __restrict__ y) {
    const int bh = blockIdx.x;
    const int b  = bh >> 5;                // / H
    const int h  = bh & 31;
    const int t  = threadIdx.x;
    const int p  = t >> 2;
    const int n0 = (t & 3) * 16;

    __shared__ float sb[NN], sc[NN], sx[PP];
    __shared__ float sdec;

    float s[16];
#pragma unroll
    for (int i = 0; i < 16; ++i) s[i] = 0.f;
    const float dp = Dmat[h * PP + p];

    for (int l = 0; l < LL; ++l) {
        const float* row = zxbca + (size_t)(b * LL + l) * PROJ;
        if (t < 64)        sb[t]        = row[2 * DI + h * NN + t];
        else if (t < 128)  sc[t - 64]   = row[2 * DI + HH * NN + h * NN + (t - 64)];
        else if (t < 192)  sx[t - 128]  = row[h * PP + (t - 128)];
        else if (t == 192) sdec         = __expf(-__expf(row[2 * DI + 2 * HH * NN + h]));
        __syncthreads();

        const float xp  = sx[p];
        const float dec = sdec;
        float acc = 0.f;
#pragma unroll
        for (int i = 0; i < 16; ++i) {
            s[i] = fmaf(s[i], dec, xp * sb[n0 + i]);
            acc  = fmaf(s[i], sc[n0 + i], acc);
        }
        acc += __shfl_xor(acc, 1, 32);
        acc += __shfl_xor(acc, 2, 32);
        if ((t & 3) == 0)
            y[(size_t)(b * LL + l) * DI + h * PP + p] = acc + xp * dp;
        __syncthreads();
    }
}

// ------------------------- SiLU gate + LayerNorm ---------------------------
// One block per row (8192 rows). Emits bf16 for GEMM2.

__global__ __launch_bounds__(256)
void gate_ln_kernel(const float* __restrict__ yscan, const float* __restrict__ zxbca,
                    const float* __restrict__ gamma, const float* __restrict__ beta,
                    uint16_t* __restrict__ out) {
    const int r = blockIdx.x;
    const int t = threadIdx.x;
    const float* yr = yscan + (size_t)r * DI;
    const float* zr = zxbca + (size_t)r * PROJ + DI;   // z_t block

    float v[8], s = 0.f, s2 = 0.f;
#pragma unroll
    for (int j = 0; j < 8; ++j) {
        int c = t + j * 256;
        float z   = zr[c];
        float val = yr[c] * (z / (1.f + __expf(-z)));  // y * silu(z)
        v[j] = val;  s += val;  s2 += val * val;
    }
#pragma unroll
    for (int o = 16; o > 0; o >>= 1) {
        s  += __shfl_xor(s,  o, 32);
        s2 += __shfl_xor(s2, o, 32);
    }
    __shared__ float rs[8], rs2[8];
    if ((t & 31) == 0) { rs[t >> 5] = s; rs2[t >> 5] = s2; }
    __syncthreads();
    float ts = 0.f, t2 = 0.f;
#pragma unroll
    for (int w = 0; w < 8; ++w) { ts += rs[w]; t2 += rs2[w]; }
    const float mu   = ts * (1.f / DI);
    const float var  = t2 * (1.f / DI) - mu * mu;
    const float rstd = rsqrtf(var + EPS);
#pragma unroll
    for (int j = 0; j < 8; ++j) {
        int c = t + j * 256;
        out[(size_t)r * DI + c] = f2bf((v[j] - mu) * rstd * gamma[c] + beta[c]);
    }
}

// ----------------------------- launcher ------------------------------------

extern "C" void kernel_launch(void* const* d_in, const int* in_sizes, int n_in,
                              void* d_out, int out_size, void* d_ws, size_t ws_size,
                              hipStream_t stream) {
    const float* x     = (const float*)d_in[0];   // [B,L,DM]
    const float* W_in  = (const float*)d_in[1];   // [PROJ,DM]
    const float* W_out = (const float*)d_in[2];   // [DM,DI]
    const float* gamma = (const float*)d_in[3];
    const float* beta  = (const float*)d_in[4];
    const float* Dmat  = (const float*)d_in[5];   // [H,P]
    float* out = (float*)d_out;
    char*  ws  = (char*)d_ws;

    // workspace layout (256B-aligned offsets)
    size_t off = 0;
    float*    zxbca  = (float*)(ws + off);    off += (size_t)MROWS * PROJ * 4;  // 269 MB
    float*    yscan  = (float*)(ws + off);    off += (size_t)MROWS * DI * 4;    //  67 MB
    uint16_t* x_bf   = (uint16_t*)(ws + off); off += (size_t)MROWS * DM * 2;    //  33 MB
    uint16_t* Win_bf = (uint16_t*)(ws + off); off += (size_t)NPAD1 * DM * 2;    //  34 MB (padded)
    uint16_t* Wout_bf= (uint16_t*)(ws + off); off += (size_t)DM * DI * 2;       //   8 MB
    uint16_t* yln_bf = (uint16_t*)(ws + off); off += (size_t)MROWS * DI * 2;    //  33 MB
    (void)ws_size; (void)in_sizes; (void)n_in; (void)out_size;

    // 1. precision conversion (x, W_in padded, W_out) to bf16
    cvt_f32_bf16<<<4096, 256, 0, stream>>>(x, x_bf,
        (long)MROWS * DM, (long)MROWS * DM);
    cvt_f32_bf16<<<4096, 256, 0, stream>>>(W_in, Win_bf,
        (long)PROJ * DM, (long)NPAD1 * DM);
    cvt_f32_bf16<<<4096, 256, 0, stream>>>(W_out, Wout_bf,
        (long)DM * DI, (long)DM * DI);

    // 2. GEMM1: zxbca[8192 x 8224] = x_bf * Win_bf^T
    wmma_gemm_bf16<<<dim3(NPAD1 / TILE_N, MROWS / TILE_M), 256, 0, stream>>>(
        x_bf, Win_bf, zxbca, DM, PROJ, PROJ);

    // 3. sequential selective scan (64 blocks = B*H)
    scan_kernel<<<BB * HH, 256, 0, stream>>>(zxbca, Dmat, yscan);

    // 4. SiLU gate + LayerNorm -> bf16
    gate_ln_kernel<<<MROWS, 256, 0, stream>>>(yscan, zxbca, gamma, beta, yln_bf);

    // 5. GEMM2: out[8192 x 2048] = yln_bf * Wout_bf^T
    wmma_gemm_bf16<<<dim3(DI / TILE_N, MROWS / TILE_M), 256, 0, stream>>>(
        yln_bf, Wout_bf, out, DI, DI, DI);
}
